// TransformerLayer_74792560492774
// MI455X (gfx1250) — matrix operand order repaired and checked by reading, not compiled, not run on hardware
//
#include <hip/hip_runtime.h>
#include <cstdint>
#include <cstddef>

#define BATCH 4
#define SEQ   4096
#define DM    128
#define DI    256
#define DSS   16
#define DC    4
#define DTR   8
#define BL    (BATCH * SEQ)   /* 16384 tokens */

typedef __bf16 bf16;
typedef bf16  bf16x8   __attribute__((ext_vector_type(8)));
typedef bf16  bf16x16  __attribute__((ext_vector_type(16)));
typedef float f32x8    __attribute__((ext_vector_type(8)));
typedef int   i32x4    __attribute__((ext_vector_type(4)));

#define AS1 __attribute__((address_space(1)))
#define AS3 __attribute__((address_space(3)))

#if __has_builtin(__builtin_amdgcn_global_load_async_to_lds_b128)
#define HAVE_ASYNC_LDS 1
#endif

__device__ __forceinline__ void wait_async0() {
#if __has_builtin(__builtin_amdgcn_s_wait_asynccnt)
  __builtin_amdgcn_s_wait_asynccnt(0);
#else
  asm volatile("s_wait_asynccnt 0x0" ::: "memory");
#endif
}

__device__ __forceinline__ bf16 f2bf(float f) {
  unsigned u = __builtin_bit_cast(unsigned, f);
  u += 0x7FFFu + ((u >> 16) & 1u);                 // round-to-nearest-even
  unsigned short h = (unsigned short)(u >> 16);
  return __builtin_bit_cast(bf16, h);
}
__device__ __forceinline__ float siluf(float x) { return x / (1.f + __expf(-x)); }
__device__ __forceinline__ float softplusf(float x) {
  return x > 20.f ? x : log1pf(__expf(x));
}

// ---------------------------------------------------------------------------
// Weight prep: f32 -> bf16 copies (row-major [N,K]) + A = -exp(A_log)
// ---------------------------------------------------------------------------
__global__ __launch_bounds__(256)
void prep_kernel(const float* __restrict__ inw, const float* __restrict__ xpw,
                 const float* __restrict__ opw, const float* __restrict__ alog,
                 bf16* __restrict__ inw_bf, bf16* __restrict__ xpw_bf,
                 bf16* __restrict__ opw_bf, float* __restrict__ aneg) {
  int i = blockIdx.x * 256 + threadIdx.x;
  if (i < 512 * 128) inw_bf[i] = f2bf(inw[i]);
  if (i < 40 * 256)  xpw_bf[i] = f2bf(xpw[i]);
  if (i < 128 * 256) opw_bf[i] = f2bf(opw[i]);
  if (i < 256 * 16)  aneg[i]   = -__expf(alog[i]);
}

// ---------------------------------------------------------------------------
// Gumbel gate + softmax + layernorm. One wave32 per token, 4 elems/lane.
// Writes xg (f32, final residual) and h (bf16, GEMM A operand).
// ---------------------------------------------------------------------------
__global__ __launch_bounds__(256)
void gate_ln_kernel(const float* __restrict__ x, const float* __restrict__ gu,
                    const float* __restrict__ nw, const float* __restrict__ nb,
                    float* __restrict__ xg, bf16* __restrict__ hbf) {
  const int t    = blockIdx.x * 8 + (threadIdx.x >> 5);
  const int lane = threadIdx.x & 31;
  const float* xp = x  + (size_t)t * DM;
  const float* gp = gu + (size_t)t * DM;
  float v[4];
  float mx = -1e30f;
#pragma unroll
  for (int i = 0; i < 4; ++i) {
    int idx = lane + 32 * i;
    float g = -__logf(-__logf(gp[idx] + 1e-20f) + 1e-20f);
    v[i] = xp[idx] + g;
    mx = fmaxf(mx, v[i]);
  }
#pragma unroll
  for (int off = 16; off >= 1; off >>= 1) mx = fmaxf(mx, __shfl_xor(mx, off));
  float sum = 0.f;
#pragma unroll
  for (int i = 0; i < 4; ++i) { v[i] = __expf(v[i] - mx); sum += v[i]; }
#pragma unroll
  for (int off = 16; off >= 1; off >>= 1) sum += __shfl_xor(sum, off);
  const float inv = 1.f / sum;
  const float mean = 1.f / (float)DM;   // softmax row sums to exactly 1
  float var = 0.f;
#pragma unroll
  for (int i = 0; i < 4; ++i) {
    v[i] *= inv;
    float d = v[i] - mean;
    var += d * d;
  }
#pragma unroll
  for (int off = 16; off >= 1; off >>= 1) var += __shfl_xor(var, off);
  var *= (1.f / (float)DM);
  const float rs = rsqrtf(var + 1e-5f);
#pragma unroll
  for (int i = 0; i < 4; ++i) {
    int idx = lane + 32 * i;
    xg[(size_t)t * DM + idx] = v[i];
    float hh = (v[i] - mean) * rs * nw[idx] + nb[idx];
    hbf[(size_t)t * DM + idx] = f2bf(hh);
  }
}

// ---------------------------------------------------------------------------
// bf16 WMMA GEMM: C[M,N] = A[M,K] * Bw[N,K]^T (+ resid), f32 accumulate.
// The 16xK A tile is shared by all 4 waves of the block: staged once into LDS
// via GLOBAL_LOAD_ASYNC_TO_LDS_B128 (ASYNCcnt) when available, else a
// cooperative b128 copy. WMMA A fragments then read as ds_load_b128.
// Operand VGPR layouts (contiguous k = 16h + s, same order for A and B):
//   A lane(m=lane&15, h=lane>>4): bf16 K run  [k0+16h, +16)
//   B lane(n=lane&15, h=lane>>4): bf16 K run  [k0+16h, +16) of weight row n
//   D lane: column n = lane&15, rows m = 8*(lane>>4) + r for VGPR r in 0..7
// PAD: mask columns >= n_valid (x_proj N=40 padded to 48).
// RESID: fused residual add on the epilogue (out_proj + xg).
// ---------------------------------------------------------------------------
template <int K, bool PAD, bool RESID>
__global__ __launch_bounds__(128)
void wmma_gemm_bf16(const bf16* __restrict__ A, const bf16* __restrict__ Bw,
                    float* __restrict__ C, const float* __restrict__ resid,
                    int n_valid, int ldc) {
  __shared__ __align__(16) bf16 atile[16 * K];
  const int tid = threadIdx.x;
  const int m0  = blockIdx.x * 16;

  // ---- stage A tile (16 rows x K bf16) into LDS ----
  constexpr int CHUNKS = (16 * K * 2) / 16;        // 16-byte chunks
  const char* gsrc = (const char*)(A + (size_t)m0 * K);
#pragma unroll
  for (int c = tid; c < CHUNKS; c += 128) {
    int off = c * 16;
#ifdef HAVE_ASYNC_LDS
    __builtin_amdgcn_global_load_async_to_lds_b128(
        (AS1 i32x4*)(gsrc + off),
        (AS3 i32x4*)((char*)atile + off), 0, 0);
#else
    *(uint4*)((char*)atile + off) = *(const uint4*)(gsrc + off);
#endif
  }
#ifdef HAVE_ASYNC_LDS
  wait_async0();
#endif
  __syncthreads();

  const int lane = tid & 31;
  const int wv   = tid >> 5;
  const int half = lane >> 4;
  const int mr   = lane & 15;
  const int n0   = (blockIdx.y * 4 + wv) * 16;
  if (PAD) {
    if (n0 >= ((n_valid + 15) & ~15)) return;      // wave-uniform
  }
  const int ncol = n0 + mr;
  int brow = ncol;
  if (PAD) brow = ncol < n_valid ? ncol : n_valid - 1;   // clamp, zero below

  const bf16* ap = atile + mr * K + 16 * half;
  const bf16* bp = Bw + (size_t)brow * K + 16 * half;
  f32x8 acc = {};
#pragma unroll
  for (int k0 = 0; k0 < K; k0 += 32) {
    bf16x8 alo = *(const bf16x8*)(ap + k0);
    bf16x8 ahi = *(const bf16x8*)(ap + k0 + 8);
    bf16x16 av = __builtin_shufflevector(alo, ahi,
        0, 1, 2, 3, 4, 5, 6, 7, 8, 9, 10, 11, 12, 13, 14, 15);
    bf16x16 bv = *(const bf16x16*)(bp + k0);
    if (PAD) {
      bf16x16 zv = {};
      bv = (ncol < n_valid) ? bv : zv;             // per-lane select, EXEC stays full
    }
    acc = __builtin_amdgcn_wmma_f32_16x16x32_bf16(
        /*neg_a=*/false, av, /*neg_b=*/false, bv,
        /*c_mod=*/(short)0, acc, /*reuse_a=*/false, /*reuse_b=*/false);
  }
  if (!PAD || ncol < n_valid) {
#pragma unroll
    for (int r = 0; r < 8; ++r) {
      int m = m0 + 8 * half + r;
      float vv = acc[r];
      if (RESID) vv += resid[(size_t)m * ldc + ncol];
      C[(size_t)m * ldc + ncol] = vv;
    }
  }
}

// ---------------------------------------------------------------------------
// Causal depthwise conv1d (width 4) + bias + SiLU on x_in = xz[:, :256].
// Writes x_ssm in both f32 (scan/skip path) and bf16 (x_proj GEMM A operand).
// ---------------------------------------------------------------------------
__global__ __launch_bounds__(256)
void conv_silu_kernel(const float* __restrict__ xz, const float* __restrict__ cw,
                      const float* __restrict__ cb,
                      float* __restrict__ xs_f, bf16* __restrict__ xs_bf) {
  size_t i = (size_t)blockIdx.x * 256 + threadIdx.x;   // over BL*DI
  int d    = (int)(i & (DI - 1));
  size_t t = i >> 8;
  int l    = (int)(t & (SEQ - 1));                     // position within batch
  float acc = cb[d];
#pragma unroll
  for (int w = 0; w < DC; ++w) {
    int ls = l - (DC - 1) + w;
    if (ls >= 0)
      acc = fmaf(cw[d * DC + w], xz[(t - (size_t)(DC - 1) + w) * (2 * DI) + d], acc);
  }
  float s = siluf(acc);
  xs_f[i]  = s;
  xs_bf[i] = f2bf(s);
}

// ---------------------------------------------------------------------------
// delta = softplus(dt @ dt_proj_w^T + dt_proj_b), dt = x_dbl[:, :8]
// ---------------------------------------------------------------------------
__global__ __launch_bounds__(256)
void delta_kernel(const float* __restrict__ x_dbl, const float* __restrict__ dtw,
                  const float* __restrict__ dtb, float* __restrict__ delta) {
  size_t i = (size_t)blockIdx.x * 256 + threadIdx.x;   // over BL*DI
  int d    = (int)(i & (DI - 1));
  size_t t = i >> 8;
  const float* dtp = x_dbl + t * 40;
  float acc = dtb[d];
#pragma unroll
  for (int r = 0; r < DTR; ++r) acc = fmaf(dtp[r], dtw[d * DTR + r], acc);
  delta[i] = softplusf(acc);
}

// ---------------------------------------------------------------------------
// Selective scan. One block = (batch b, 16 channels); thread owns one (d,s).
// 16-lane butterfly reduction over the state dim via __shfl_xor (wave32).
// Serial over L=4096 — latency critical path; prefetch 8 steps ahead.
// ---------------------------------------------------------------------------
__global__ __launch_bounds__(256)
void scan_kernel(const float* __restrict__ delta, const float* __restrict__ x_dbl,
                 const float* __restrict__ xs, const float* __restrict__ aneg,
                 float* __restrict__ y) {
  const int b      = blockIdx.x >> 4;
  const int dchunk = blockIdx.x & 15;
  const int d = dchunk * 16 + (threadIdx.x >> 4);
  const int s = threadIdx.x & 15;
  const float a = aneg[d * DSS + s];
  float hc = 0.f;
  const size_t tbase = (size_t)b * SEQ;
  for (int l = 0; l < SEQ; ++l) {
    size_t t = tbase + l;
    float dl = delta[t * DI + d];
    float bm = x_dbl[t * 40 + DTR + s];
    float cm = x_dbl[t * 40 + DTR + DSS + s];
    float xv = xs[t * DI + d];
    hc = __expf(dl * a) * hc + dl * bm * xv;
    float p = hc * cm;
    p += __shfl_xor(p, 1, 16);
    p += __shfl_xor(p, 2, 16);
    p += __shfl_xor(p, 4, 16);
    p += __shfl_xor(p, 8, 16);
    if (s == 0) y[t * DI + d] = p;
    if ((l & 7) == 0 && l + 8 < SEQ) {
      __builtin_prefetch(&delta[(t + 8) * DI + d], 0, 1);
      __builtin_prefetch(&x_dbl[(t + 8) * 40 + s], 0, 1);
      __builtin_prefetch(&xs[(t + 8) * DI + d], 0, 1);
    }
  }
}

// ---------------------------------------------------------------------------
// y = (y + x_ssm * D) * silu(z)  -> bf16 for out_proj GEMM A operand
// ---------------------------------------------------------------------------
__global__ __launch_bounds__(256)
void ygate_kernel(const float* __restrict__ y, const float* __restrict__ xs_f,
                  const float* __restrict__ xz, const float* __restrict__ Dp,
                  bf16* __restrict__ yy_bf) {
  size_t i = (size_t)blockIdx.x * 256 + threadIdx.x;   // over BL*DI
  int d    = (int)(i & (DI - 1));
  size_t t = i >> 8;
  float z  = xz[t * (2 * DI) + DI + d];
  float vv = (y[i] + xs_f[i] * Dp[d]) * siluf(z);
  yy_bf[i] = f2bf(vv);
}

// ---------------------------------------------------------------------------
extern "C" void kernel_launch(void* const* d_in, const int* in_sizes, int n_in,
                              void* d_out, int out_size, void* d_ws, size_t ws_size,
                              hipStream_t stream) {
  (void)in_sizes; (void)n_in; (void)out_size; (void)ws_size;
  const float* x    = (const float*)d_in[0];
  const float* gu   = (const float*)d_in[1];
  const float* nw   = (const float*)d_in[2];
  const float* nb   = (const float*)d_in[3];
  const float* inw  = (const float*)d_in[4];
  const float* cw   = (const float*)d_in[5];
  const float* cb   = (const float*)d_in[6];
  const float* xpw  = (const float*)d_in[7];
  const float* dtw  = (const float*)d_in[8];
  const float* dtb  = (const float*)d_in[9];
  const float* alog = (const float*)d_in[10];
  const float* Dp   = (const float*)d_in[11];
  const float* opw  = (const float*)d_in[12];
  float* out = (float*)d_out;

  char* p = (char*)d_ws;
  auto alloc = [&](size_t bytes) -> char* {
    char* r = p;
    p += (bytes + 255) & ~(size_t)255;
    return r;
  };
  float* xg     = (float*)alloc((size_t)BL * DM * 4);
  bf16*  hbf    = (bf16*) alloc((size_t)BL * DM * 2);
  bf16*  inw_bf = (bf16*) alloc((size_t)512 * 128 * 2);
  bf16*  xpw_bf = (bf16*) alloc((size_t)40 * 256 * 2);
  bf16*  opw_bf = (bf16*) alloc((size_t)128 * 256 * 2);
  float* aneg   = (float*)alloc((size_t)256 * 16 * 4);
  float* xz     = (float*)alloc((size_t)BL * 512 * 4);
  float* xs_f   = (float*)alloc((size_t)BL * DI * 4);
  bf16*  xs_bf  = (bf16*) alloc((size_t)BL * DI * 2);
  float* x_dbl  = (float*)alloc((size_t)BL * 40 * 4);
  float* delta  = (float*)alloc((size_t)BL * DI * 4);
  float* yv     = (float*)alloc((size_t)BL * DI * 4);
  bf16*  yy_bf  = (bf16*) alloc((size_t)BL * DI * 2);

  prep_kernel<<<256, 256, 0, stream>>>(inw, xpw, opw, alog, inw_bf, xpw_bf, opw_bf, aneg);
  gate_ln_kernel<<<BL / 8, 256, 0, stream>>>(x, gu, nw, nb, xg, hbf);
  // in_proj: [BL,128] x [512,128]^T -> xz [BL,512]
  wmma_gemm_bf16<128, false, false><<<dim3(BL / 16, 8), 128, 0, stream>>>(
      hbf, inw_bf, xz, nullptr, 512, 512);
  conv_silu_kernel<<<(BL * DI) / 256, 256, 0, stream>>>(xz, cw, cb, xs_f, xs_bf);
  // x_proj: [BL,256] x [40,256]^T -> x_dbl [BL,40]  (N padded to 48 in-wave)
  wmma_gemm_bf16<256, true, false><<<dim3(BL / 16, 1), 128, 0, stream>>>(
      xs_bf, xpw_bf, x_dbl, nullptr, 40, 40);
  delta_kernel<<<(BL * DI) / 256, 256, 0, stream>>>(x_dbl, dtw, dtb, delta);
  scan_kernel<<<BATCH * 16, 256, 0, stream>>>(delta, x_dbl, xs_f, aneg, yv);
  ygate_kernel<<<(BL * DI) / 256, 256, 0, stream>>>(yv, xs_f, xz, Dp, yy_bf);
  // out_proj + residual xg: [BL,256] x [128,256]^T + xg -> out [BL,128]
  wmma_gemm_bf16<256, false, true><<<dim3(BL / 16, 2), 128, 0, stream>>>(
      yy_bf, opw_bf, out, xg, 128, 128);
}
